// VectorQuantizer_14448269984284
// MI455X (gfx1250) — matrix operand order repaired and checked by reading, not compiled, not run on hardware
//
#include <hip/hip_runtime.h>
#include <stdint.h>

#define NROWS   32768
#define NCODES  4096
#define DIM     512
#define APITCH  520                       // bf16 elems per LDS row (512 + 8 pad -> conflict-free b128)
#define WG_ROWS 64
#define LDS_BYTES (2u * WG_ROWS * APITCH * 2u)   // hi+lo planes = 133,120 B (<320KB WGP LDS)

typedef __attribute__((ext_vector_type(16))) __bf16        v16bf;
typedef __attribute__((ext_vector_type(8)))  __bf16        v8bf;
typedef __attribute__((ext_vector_type(8)))  unsigned short v8us;
typedef __attribute__((ext_vector_type(4)))  unsigned short v4us;
typedef __attribute__((ext_vector_type(8)))  float         v8f;
typedef __attribute__((ext_vector_type(4)))  float         v4f;

static __device__ __forceinline__ unsigned short bf16_rne(float f) {
  unsigned u = __float_as_uint(f);
  u += 0x7FFFu + ((u >> 16) & 1u);
  return (unsigned short)(u >> 16);
}
static __device__ __forceinline__ float bf16_f32(unsigned short h) {
  return __uint_as_float((unsigned)h << 16);
}
static __device__ __forceinline__ v16bf frag_cat(v8us a, v8us b) {
  v8bf x = __builtin_bit_cast(v8bf, a);
  v8bf y = __builtin_bit_cast(v8bf, b);
  return __builtin_shufflevector(x, y, 0,1,2,3,4,5,6,7,8,9,10,11,12,13,14,15);
}
static __device__ __forceinline__ v8us ldv8(const unsigned short* p) {
  return *(const v8us*)p;
}

// ---------------- Kernel 1: split embed into bf16 hi/lo planes + z2 ----------------
__global__ __launch_bounds__(256) void vq_prep_kernel(
    const float* __restrict__ e,
    unsigned short* __restrict__ ehi,
    unsigned short* __restrict__ elo,
    float* __restrict__ z2) {
  __shared__ float red[256];
  const int c = blockIdx.x;
  const int t = threadIdx.x;
  const float* row = e + (size_t)c * DIM;
  float ss = 0.0f;
  for (int k = t; k < DIM; k += 256) {
    const float v = row[k];
    const unsigned short h = bf16_rne(v);
    ehi[(size_t)c * DIM + k] = h;
    elo[(size_t)c * DIM + k] = bf16_rne(v - bf16_f32(h));
    ss += v * v;
  }
  red[t] = ss;
  __syncthreads();
  for (int s = 128; s > 0; s >>= 1) {
    if (t < s) red[t] += red[t + s];
    __syncthreads();
  }
  if (t == 0) z2[c] = red[0];
}

// ---------------- Kernel 2: fused (split-bf16 WMMA GEMM) + running argmin ----------------
__global__ __launch_bounds__(256) void vq_argmin_kernel(
    const float* __restrict__ x,
    const unsigned short* __restrict__ ehi,
    const unsigned short* __restrict__ elo,
    const float* __restrict__ z2,
    int* __restrict__ idxOut) {
  extern __shared__ unsigned short lds[];
  unsigned short* Ahi = lds;
  unsigned short* Alo = lds + WG_ROWS * APITCH;

  const int tid = threadIdx.x;
  const int rowBase = blockIdx.x * WG_ROWS;

  // Stage 64 rows of x -> bf16 hi/lo planes in LDS (converted once, reused for all 4096 codes)
  for (int i = tid * 4; i < WG_ROWS * DIM; i += 256 * 4) {
    const int r = i >> 9;
    const int c = i & (DIM - 1);
    v4f v = *(const v4f*)(x + (size_t)(rowBase + r) * DIM + c);
    v4us h, l;
#pragma unroll
    for (int j = 0; j < 4; ++j) {
      const unsigned short hb = bf16_rne(v[j]);
      h[j] = hb;
      l[j] = bf16_rne(v[j] - bf16_f32(hb));
    }
    *(v4us*)&Ahi[r * APITCH + c] = h;
    *(v4us*)&Alo[r * APITCH + c] = l;
  }
  __syncthreads();

  const int wave   = tid >> 5;
  const int lane   = tid & 31;
  const int laneLo = lane & 15;
  const int laneHi = lane >> 4;

  float minv[4][8];
  int   mini[4][8];
#pragma unroll
  for (int m = 0; m < 4; ++m)
#pragma unroll
    for (int i = 0; i < 8; ++i) { minv[m][i] = 3.402823466e38f; mini[m][i] = 0; }

  // Two INDEPENDENT per-lane plane bases. The lo-plane offset is laundered through
  // an empty asm constraint so the compiler cannot fold it back into the hi base
  // (lo plane sits at +66,560B, beyond the 16-bit DS immediate). With two opaque
// bases, every in-loop ds_load_b128 offset is m*16640 + k0*2 (+16) <= 50,896:
  // immediate-only addressing, zero in-loop VALU address math, no WAR-hazard nops.
  const unsigned laneAOff = (unsigned)(laneLo * APITCH + laneHi * 16);
  unsigned loOff = (unsigned)(WG_ROWS * APITCH) + laneAOff;
  asm("" : "+v"(loOff));
  const unsigned short* __restrict__ AhiL = Ahi + laneAOff;
  const unsigned short* __restrict__ AloL = lds + loOff;

  const int codeBase = wave * (NCODES / 8);   // 8 waves split the codebook

  for (int it = 0; it < NCODES / 8 / 32; ++it) {
    const int c0 = codeBase + it * 32;
    // B fragment lane pointers (contiguous k order, same as A: lanes<16 hold K 0..15);
    // n=1 subtile and k offsets are immediates off these two bases.
    const size_t boff0 = (size_t)(c0 + laneLo) * DIM + (size_t)laneHi * 16;
    const unsigned short* __restrict__ bh = ehi + boff0;
    const unsigned short* __restrict__ bl = elo + boff0;

    v8f acc[4][2] = {};

#pragma unroll 2
    for (int kt = 0; kt < DIM / 32; ++kt) {
      const int k0 = kt * 32;
      v16bf bhi[2], blo[2];
      bhi[0] = frag_cat(ldv8(bh + k0),            ldv8(bh + k0 + 8));
      blo[0] = frag_cat(ldv8(bl + k0),            ldv8(bl + k0 + 8));
      bhi[1] = frag_cat(ldv8(bh + 16 * DIM + k0), ldv8(bh + 16 * DIM + k0 + 8));
      blo[1] = frag_cat(ldv8(bl + 16 * DIM + k0), ldv8(bl + 16 * DIM + k0 + 8));
#pragma unroll
      for (int m = 0; m < 4; ++m) {
        const int aoff = m * 16 * APITCH + k0;   // compile-time constant per unrolled m
        const v16bf ahi = frag_cat(ldv8(AhiL + aoff), ldv8(AhiL + aoff + 8));
        const v16bf alo = frag_cat(ldv8(AloL + aoff), ldv8(AloL + aoff + 8));
#pragma unroll
        for (int n = 0; n < 2; ++n) {
          // 3-term split: hi*hi + lo*hi + hi*lo  (f32-class accuracy)
          acc[m][n] = __builtin_amdgcn_wmma_f32_16x16x32_bf16(
              false, ahi, false, bhi[n], (short)0, acc[m][n], false, false);
          acc[m][n] = __builtin_amdgcn_wmma_f32_16x16x32_bf16(
              false, alo, false, bhi[n], (short)0, acc[m][n], false, false);
          acc[m][n] = __builtin_amdgcn_wmma_f32_16x16x32_bf16(
              false, ahi, false, blo[n], (short)0, acc[m][n], false, false);
        }
      }
    }

    // score = z2[j] - 2*x.z  (x2 is constant per row -> irrelevant for argmin)
#pragma unroll
    for (int n = 0; n < 2; ++n) {
      const int code = c0 + n * 16 + laneLo;
      const float zz = z2[code];
#pragma unroll
      for (int m = 0; m < 4; ++m)
#pragma unroll
        for (int i = 0; i < 8; ++i) {
          const float s = fmaf(-2.0f, acc[m][n][i], zz);
          const bool b = s < minv[m][i];           // strict < keeps first occurrence
          minv[m][i] = b ? s : minv[m][i];
          mini[m][i] = b ? code : mini[m][i];
        }
    }
  }

  // Cross-lane / cross-wave argmin reduction (reuse A plane as scratch after barrier)
  __syncthreads();
  uint2* scratch = (uint2*)lds;   // [WG_ROWS][128] (val,idx) pairs = 64KB
#pragma unroll
  for (int m = 0; m < 4; ++m)
#pragma unroll
    for (int i = 0; i < 8; ++i) {
      const int row = m * 16 + laneHi * 8 + i;    // ISA f32 C/D layout
      scratch[row * 128 + wave * 16 + laneLo] =
          make_uint2(__float_as_uint(minv[m][i]), (unsigned)mini[m][i]);
    }
  __syncthreads();

  if (tid < WG_ROWS) {
    float bv = 3.402823466e38f;
    int   bi = 0x7FFFFFFF;
    for (int s = 0; s < 128; ++s) {
      const uint2 p = scratch[tid * 128 + s];
      const float v = __uint_as_float(p.x);
      const int  ci = (int)p.y;
      if (v < bv || (v == bv && ci < bi)) { bv = v; bi = ci; }   // jnp.argmin tie-break
    }
    idxOut[rowBase + tid] = bi;
  }
}

// ---------------- Kernel 3: gather + STE output + per-row squared error ----------------
__global__ __launch_bounds__(256) void vq_gather_kernel(
    const float* __restrict__ x,
    const float* __restrict__ e,
    const int* __restrict__ idx,
    float* __restrict__ out,
    float* __restrict__ partial) {
  __shared__ float red[256];
  const int row = blockIdx.x;
  const int t = threadIdx.x;
  const int code = idx[row];
  const float* xr = x + (size_t)row * DIM;
  const float* er = e + (size_t)code * DIM;
  float ss = 0.0f;
  for (int k = t; k < DIM; k += 256) {
    const float xv = xr[k];
    const float ev = er[k];
    const float d = xv - ev;
    out[(size_t)row * DIM + k] = xv - d;   // == x + (vq - x), bit-exact STE
    ss += d * d;
  }
  red[t] = ss;
  __syncthreads();
  for (int s = 128; s > 0; s >>= 1) {
    if (t < s) red[t] += red[t + s];
    __syncthreads();
  }
  if (t == 0) partial[row] = red[0];
}

// ---------------- Kernel 4: deterministic final loss reduction ----------------
__global__ __launch_bounds__(256) void vq_loss_kernel(
    const float* __restrict__ partial, float* __restrict__ out_loss) {
  __shared__ float red[256];
  const int t = threadIdx.x;
  float s = 0.0f;
  for (int i = t; i < NROWS; i += 256) s += partial[i];
  red[t] = s;
  __syncthreads();
  for (int w = 128; w > 0; w >>= 1) {
    if (t < w) red[t] += red[t + w];
    __syncthreads();
  }
  // loss = BETA*mse + GAMMA*mse = 1.25 * sum / (N*DIM);  N*DIM = 2^24
  if (t == 0) out_loss[0] = 1.25f * red[0] * (1.0f / 16777216.0f);
}

extern "C" void kernel_launch(void* const* d_in, const int* in_sizes, int n_in,
                              void* d_out, int out_size, void* d_ws, size_t ws_size,
                              hipStream_t stream) {
  (void)in_sizes; (void)n_in; (void)out_size; (void)ws_size;
  const float* x = (const float*)d_in[0];
  const float* e = (const float*)d_in[1];

  char* ws = (char*)d_ws;                                   // ~8.7 MB used
  unsigned short* ehi = (unsigned short*)ws;                              // 4 MiB
  unsigned short* elo = (unsigned short*)(ws + (size_t)NCODES * DIM * 2); // 4 MiB
  float* z2      = (float*)(ws + (size_t)NCODES * DIM * 4);               // 16 KiB
  int*   idx     = (int*)((char*)z2 + (size_t)NCODES * 4);                // 128 KiB
  float* partial = (float*)((char*)idx + (size_t)NROWS * 4);              // 128 KiB

  float* out = (float*)d_out;
  float* out_loss = out + (size_t)NROWS * DIM;

  hipFuncSetAttribute(reinterpret_cast<const void*>(vq_argmin_kernel),
                      hipFuncAttributeMaxDynamicSharedMemorySize, (int)LDS_BYTES);

  vq_prep_kernel<<<NCODES, 256, 0, stream>>>(e, ehi, elo, z2);
  vq_argmin_kernel<<<NROWS / WG_ROWS, 256, LDS_BYTES, stream>>>(x, ehi, elo, z2, idx);
  vq_gather_kernel<<<NROWS, 256, 0, stream>>>(x, e, idx, out, partial);
  vq_loss_kernel<<<1, 256, 0, stream>>>(partial, out_loss);
}
